// BlockGNNComposer_40200893891314
// MI455X (gfx1250) — matrix operand order repaired and checked by reading, not compiled, not run on hardware
//
#include <hip/hip_runtime.h>
#include <stdint.h>

typedef __attribute__((ext_vector_type(16))) __bf16 v16bf;
typedef __attribute__((ext_vector_type(8)))  float  v8f;

#define KNODES 2048
#define NEDGE  16384
#define NCONS  200000
#define NVARS  400000

// ---------------- helpers ----------------
__device__ inline unsigned fkey(float f) {            // order-preserving float->uint
  unsigned u = __float_as_uint(f);
  return (u & 0x80000000u) ? ~u : (u | 0x80000000u);
}
__device__ inline float funkey(unsigned k) {
  unsigned u = (k & 0x80000000u) ? (k & 0x7FFFFFFFu) : ~k;
  return __uint_as_float(u);
}

// Single-tile accumulate: C += A(16xK)*B(Kx16), bf16 WMMA over `ksteps` K=32 chunks.
// A frag (ISA 7.12.2): lane m=l&15, half=l>>4 -> dwords {0..3,8..11}+4*half per chunk.
// B frag (ISA 7.12.5): lane n=l&15, half=l>>4 -> dwords {0..3,8..11}+4*half per chunk.
__device__ inline v8f wmma_acc(const uint32_t* a_lds, int aStrideDW,
                               const uint32_t* b_col, int ksteps,
                               int lane, v8f c) {
  const int m = lane & 15;
  const int half = lane >> 4;
  const uint32_t* ar = a_lds + m * aStrideDW + half * 4;
  const uint32_t* br = b_col + half * 4;
  for (int ks = 0; ks < ksteps; ++ks) {
    union { v16bf v; uint32_t u[8]; } A, B;
    const int kb = ks * 16;
#pragma unroll
    for (int p = 0; p < 4; ++p) {
      A.u[p]     = ar[kb + p];
      A.u[p + 4] = ar[kb + 8 + p];
    }
#pragma unroll
    for (int p = 0; p < 4; ++p) { B.u[p] = br[kb + p]; B.u[p + 4] = br[kb + 8 + p]; }
    c = __builtin_amdgcn_wmma_f32_16x16x32_bf16(false, A.v, false, B.v, (short)0, c, false, false);
  }
  return c;
}

// Dual-M-tile accumulate: load each B fragment ONCE, apply to two 16-row A tiles.
// Halves weight (L2) traffic: 1x global_load_b128 pair per 2 WMMAs.
__device__ inline void wmma_acc2(const uint32_t* a0_lds, const uint32_t* a1_lds,
                                 int aStrideDW, const uint32_t* b_col, int ksteps,
                                 int lane, v8f& c0, v8f& c1) {
  const int m = lane & 15;
  const int half = lane >> 4;
  const uint32_t* ar0 = a0_lds + m * aStrideDW + half * 4;
  const uint32_t* ar1 = a1_lds + m * aStrideDW + half * 4;
  const uint32_t* br = b_col + half * 4;
  for (int ks = 0; ks < ksteps; ++ks) {
    union { v16bf v; uint32_t u[8]; } A0, A1, B;
    const int kb = ks * 16;
#pragma unroll
    for (int p = 0; p < 4; ++p) { B.u[p] = br[kb + p]; B.u[p + 4] = br[kb + 8 + p]; }
#pragma unroll
    for (int p = 0; p < 4; ++p) {
      A0.u[p]     = ar0[kb + p];
      A0.u[p + 4] = ar0[kb + 8 + p];
      A1.u[p]     = ar1[kb + p];
      A1.u[p + 4] = ar1[kb + 8 + p];
    }
    c0 = __builtin_amdgcn_wmma_f32_16x16x32_bf16(false, A0.v, false, B.v, (short)0, c0, false, false);
    c1 = __builtin_amdgcn_wmma_f32_16x16x32_bf16(false, A1.v, false, B.v, (short)0, c1, false, false);
  }
}

// ---------------- weight prep: f32 [K,N] -> bf16 [N,Kp] (zero-padded K) ----------------
__global__ void wtrans_kernel(const float* __restrict__ W, __bf16* __restrict__ Wt,
                              int K, int N, int Kp) {
  int idx = blockIdx.x * 256 + threadIdx.x;
  if (idx >= N * Kp) return;
  int n = idx / Kp, k = idx - n * Kp;
  float v = (k < K) ? W[(size_t)k * N + n] : 0.f;
  Wt[idx] = (__bf16)v;
}

// ---------------- generic 16-row WMMA GEMM: out = act( A[M,K] @ W[K,N] + bias ) ----------------
__global__ __launch_bounds__(32) void gemm_rows16(
    const float* __restrict__ A, int M, int K,
    const uint32_t* __restrict__ Wt, int Kp,
    const float* __restrict__ bias, float* __restrict__ out, int N, int act) {
  __shared__ __bf16 sA[16 * 256];
  const int lane = threadIdx.x;
  const int m0 = blockIdx.x * 16;
  for (int idx = lane; idx < 16 * Kp; idx += 32) {
    int r = idx / Kp, k = idx - r * Kp;
    int row = m0 + r;
    float v = (row < M && k < K) ? A[(size_t)row * K + k] : 0.f;
    sA[r * Kp + k] = (__bf16)v;
  }
  __syncthreads();
  const uint32_t* aw = (const uint32_t*)sA;
  const int KpDW = Kp >> 1;
  const int ksteps = Kp >> 5;
  const int n = lane & 15, half = lane >> 4;
  for (int n0 = 0; n0 < N; n0 += 16) {
    const int col = n0 + n;
    v8f c = {};
    c = wmma_acc(aw, KpDW, Wt + (size_t)col * KpDW, ksteps, lane, c);
    const float bb = bias ? bias[col] : 0.f;
#pragma unroll
    for (int j = 0; j < 8; ++j) {
      int row = m0 + j + 8 * half;
      if (row < M) {
        float v = c[j] + bb;
        if (act) v = fmaxf(v, 0.f);
        out[(size_t)row * N + col] = v;
      }
    }
  }
}

// ---------------- LayerNorm over D=32 rows (input embed) ----------------
__global__ __launch_bounds__(32) void ln32_kernel(const float* __restrict__ in,
                                                  const float* __restrict__ g,
                                                  const float* __restrict__ b,
                                                  float* __restrict__ out) {
  const int row = blockIdx.x, t = threadIdx.x;
  float v = in[(size_t)row * 32 + t];
  float s = v, ss = v * v;
  for (int m = 16; m >= 1; m >>= 1) { s += __shfl_xor(s, m, 32); ss += __shfl_xor(ss, m, 32); }
  float mu = s * (1.f / 32.f);
  float var = ss * (1.f / 32.f) - mu * mu;
  float rs = rsqrtf(var + 1e-5f);
  out[(size_t)row * 32 + t] = (v - mu) * rs * g[t] + b[t];
}

// ---------------- LN( agg + conv_bias + residual ) over D=128 ----------------
__global__ __launch_bounds__(128) void ln_resid_kernel(
    const float* __restrict__ agg, const float* __restrict__ bias,
    const float* __restrict__ resid, const float* __restrict__ g,
    const float* __restrict__ b, float* __restrict__ out) {
  __shared__ float sp[8];
  const int row = blockIdx.x, t = threadIdx.x;
  float v = agg[(size_t)row * 128 + t] + bias[t] + resid[(size_t)row * 128 + t];
  float s = v, ss = v * v;
  for (int m = 16; m >= 1; m >>= 1) { s += __shfl_xor(s, m, 32); ss += __shfl_xor(ss, m, 32); }
  const int wv = t >> 5;
  if ((t & 31) == 0) { sp[wv] = s; sp[4 + wv] = ss; }
  __syncthreads();
  float S = sp[0] + sp[1] + sp[2] + sp[3];
  float SS = sp[4] + sp[5] + sp[6] + sp[7];
  float mu = S * (1.f / 128.f);
  float var = SS * (1.f / 128.f) - mu * mu;
  float rs = rsqrtf(var + 1e-5f);
  out[(size_t)row * 128 + t] = (v - mu) * rs * g[t] + b[t];
}

// ---------------- GATv2 scatter phase ----------------
// one wave per (directed edge, head): logit + atomic max
__global__ __launch_bounds__(256) void edge_logit_kernel(
    const int* __restrict__ ei, const float* __restrict__ xl, const float* __restrict__ xr,
    const float* __restrict__ ef, const float* __restrict__ att,
    float* __restrict__ logitb, unsigned* __restrict__ mxkey) {
  const int gid = blockIdx.x * 256 + threadIdx.x;
  const int w = gid >> 5, lane = gid & 31;
  const int e2 = w >> 2, h = w & 3;
  if (e2 >= 2 * NEDGE) return;
  const int e = (e2 < NEDGE) ? e2 : e2 - NEDGE;
  const int src = (e2 < NEDGE) ? ei[e] : ei[NEDGE + e];
  const int dst = (e2 < NEDGE) ? ei[NEDGE + e] : ei[e];
  float m = xl[(size_t)src * 128 + h * 32 + lane] +
            xr[(size_t)dst * 128 + h * 32 + lane] +
            ef[(size_t)e * 128 + h * 32 + lane];
  m = (m > 0.f) ? m : 0.2f * m;                    // leaky_relu(0.2)
  float t = m * att[h * 32 + lane];
  for (int d = 16; d >= 1; d >>= 1) t += __shfl_xor(t, d, 32);
  if (lane == 0) {
    logitb[(size_t)e2 * 4 + h] = t;
    atomicMax(&mxkey[dst * 4 + h], fkey(t));
  }
}

__global__ __launch_bounds__(256) void edge_exp_kernel(
    const int* __restrict__ ei, const float* __restrict__ logitb,
    const unsigned* __restrict__ mxkey, float* __restrict__ exb, float* __restrict__ den) {
  const int i = blockIdx.x * 256 + threadIdx.x;
  if (i >= 2 * NEDGE * 4) return;
  const int e2 = i >> 2, h = i & 3;
  const int e = (e2 < NEDGE) ? e2 : e2 - NEDGE;
  const int dst = (e2 < NEDGE) ? ei[NEDGE + e] : ei[e];
  float mx = funkey(mxkey[dst * 4 + h]);
  float ex = expf(logitb[i] - mx);
  exb[i] = ex;
  atomicAdd(&den[dst * 4 + h], ex);
}

__global__ __launch_bounds__(256) void edge_agg_kernel(
    const int* __restrict__ ei, const float* __restrict__ exb, const float* __restrict__ den,
    const float* __restrict__ xl, float* __restrict__ agg) {
  const int i = blockIdx.x * 256 + threadIdx.x;
  if (i >= 2 * NEDGE * 128) return;
  const int e2 = i >> 7, c = i & 127, h = c >> 5;
  const int e = (e2 < NEDGE) ? e2 : e2 - NEDGE;
  const int src = (e2 < NEDGE) ? ei[e] : ei[NEDGE + e];
  const int dst = (e2 < NEDGE) ? ei[NEDGE + e] : ei[e];
  const float alpha = exb[(size_t)e2 * 4 + h] / (den[dst * 4 + h] + 1e-16f);
  atomicAdd(&agg[(size_t)dst * 128 + c], alpha * xl[(size_t)src * 128 + c]);
}

// ---------------- fused composer MLP (208->256 LN ReLU ->256 LN ReLU ->64) ----------------
// One wave normalizes 16 rows (rows spread over lane&15, halves over lane>>4).
__device__ inline void ln_relu_wave(const float* __restrict__ hb, __bf16* __restrict__ act,
                                    const float* __restrict__ g, const float* __restrict__ bt,
                                    int lane) {
  const int r = lane & 15, part = lane >> 4;
  const float* row = hb + r * 256 + part * 128;
  float s = 0.f, ss = 0.f;
  for (int k = 0; k < 128; ++k) { float v = row[k]; s += v; ss += v * v; }
  s  += __shfl_xor(s, 16, 32);
  ss += __shfl_xor(ss, 16, 32);
  const float mu = s * (1.f / 256.f);
  const float var = ss * (1.f / 256.f) - mu * mu;
  const float rs = rsqrtf(var + 1e-5f);
  __bf16* arow = act + r * 256 + part * 128;
  const float* gg = g + part * 128;
  const float* bb = bt + part * 128;
  for (int k = 0; k < 128; ++k) {
    float v = (row[k] - mu) * rs * gg[k] + bb[k];
    arow[k] = (__bf16)fmaxf(v, 0.f);
  }
}

// Block = 2 waves x 32 rows each (two 16-row WMMA sub-tiles per wave, B reused 2x).
__global__ __launch_bounds__(64) void composer_kernel(
    int R,
    const float* __restrict__ sub, const int* __restrict__ bid,
    const float* __restrict__ bnd, const float* __restrict__ z,
    const uint32_t* __restrict__ W1t, const float* __restrict__ b1,
    const float* __restrict__ g1, const float* __restrict__ bt1,
    const uint32_t* __restrict__ W2t, const float* __restrict__ b2,
    const float* __restrict__ g2, const float* __restrict__ bt2,
    const uint32_t* __restrict__ W3t, const float* __restrict__ b3,
    float* __restrict__ out) {
  extern __shared__ char smem[];
  const int w = threadIdx.x >> 5;
  const int lane = threadIdx.x & 31;
  __bf16* act = (__bf16*)smem + (size_t)w * 32 * 256;                       // 2 waves * 16KB
  float*  hb  = (float*)(smem + 2 * 32 * 256 * 2) + (size_t)w * 32 * 256;   // 2 waves * 32KB
  const int rowbase = blockIdx.x * 64 + w * 32;

  // stage concat([sub(64), z[bid](128), bnd(16), zeros]) as bf16 rows of 256
  for (int idx = lane; idx < 32 * 256; idx += 32) {
    const int r = idx >> 8, k = idx & 255;
    const int grow = rowbase + r;
    float v = 0.f;
    if (grow < R) {
      if (k < 64)       v = sub[(size_t)grow * 64 + k];
      else if (k < 192) v = z[(size_t)bid[grow] * 128 + (k - 64)];
      else if (k < 208) v = bnd[(size_t)grow * 16 + (k - 192)];
    }
    act[idx] = (__bf16)v;
  }
  __syncthreads();

  const uint32_t* aw0 = (const uint32_t*)act;                 // rows  0..15
  const uint32_t* aw1 = (const uint32_t*)(act + 16 * 256);    // rows 16..31
  const int n = lane & 15, half = lane >> 4;

  // layer 1: K=208 (padded 224 -> 7 ksteps), W1t rows are 224 bf16 = 112 dwords
  for (int t = 0; t < 16; ++t) {
    const int col = t * 16 + n;
    v8f c0 = {}, c1 = {};
    wmma_acc2(aw0, aw1, 128, W1t + (size_t)col * 112, 7, lane, c0, c1);
    const float bb = b1[col];
#pragma unroll
    for (int j = 0; j < 8; ++j) {
      hb[(j + 8 * half) * 256 + col]        = c0[j] + bb;
      hb[(16 + j + 8 * half) * 256 + col]   = c1[j] + bb;
    }
  }
  __syncthreads();
  ln_relu_wave(hb, act, g1, bt1, lane);
  ln_relu_wave(hb + 16 * 256, act + 16 * 256, g1, bt1, lane);
  __syncthreads();

  // layer 2: K=256 (8 ksteps)
  for (int t = 0; t < 16; ++t) {
    const int col = t * 16 + n;
    v8f c0 = {}, c1 = {};
    wmma_acc2(aw0, aw1, 128, W2t + (size_t)col * 128, 8, lane, c0, c1);
    const float bb = b2[col];
#pragma unroll
    for (int j = 0; j < 8; ++j) {
      hb[(j + 8 * half) * 256 + col]        = c0[j] + bb;
      hb[(16 + j + 8 * half) * 256 + col]   = c1[j] + bb;
    }
  }
  __syncthreads();
  ln_relu_wave(hb, act, g2, bt2, lane);
  ln_relu_wave(hb + 16 * 256, act + 16 * 256, g2, bt2, lane);
  __syncthreads();

  // layer 3: 256 -> 64, write straight to global
  for (int t = 0; t < 4; ++t) {
    const int col = t * 16 + n;
    v8f c0 = {}, c1 = {};
    wmma_acc2(aw0, aw1, 128, W3t + (size_t)col * 128, 8, lane, c0, c1);
    const float bb = b3[col];
#pragma unroll
    for (int j = 0; j < 8; ++j) {
      const int r0 = rowbase + j + 8 * half;
      const int r1 = r0 + 16;
      if (r0 < R) out[(size_t)r0 * 64 + col] = c0[j] + bb;
      if (r1 < R) out[(size_t)r1 * 64 + col] = c1[j] + bb;
    }
  }
}

// ---------------- host ----------------
extern "C" void kernel_launch(void* const* d_in, const int* in_sizes, int n_in,
                              void* d_out, int out_size, void* d_ws, size_t ws_size,
                              hipStream_t stream) {
  (void)in_sizes; (void)out_size; (void)ws_size;
  if (n_in < 57) return;

  const float* block_features = (const float*)d_in[0];
  const int*   ei             = (const int*)d_in[1];   // [2,E] row-major
  const float* edge_attr      = (const float*)d_in[2];
  const int*   cons_bid       = (const int*)d_in[3];
  const int*   vars_bid       = (const int*)d_in[4];
  const float* c_sub          = (const float*)d_in[5];
  const float* v_sub          = (const float*)d_in[6];
  const float* c_bnd          = (const float*)d_in[7];
  const float* v_bnd          = (const float*)d_in[8];
  const float* P[48];
  for (int i = 0; i < 48; ++i) P[i] = (const float*)d_in[9 + i];
  // params pytree flattened with sorted keys:
  enum { pc1_We = 0, pc1_Wl, pc1_Wr, pc1_att, pc1_bias, pc1_bl, pc1_br,
         pc2_We, pc2_Wl, pc2_Wr, pc2_att, pc2_bias, pc2_bl, pc2_br,
         pcons_W1, pcons_W2, pcons_W3, pcons_b1, pcons_b2, pcons_b3,
         pcons_bt1, pcons_bt2, pcons_g1, pcons_g2,
         pedge_W, pedge_b, pin_W, pin_b, pin_ln_b, pin_ln_g,
         pn1_b, pn1_g, pn2_b, pn2_g,
         pout_W1, pout_W2, pout_b1, pout_b2,
         pvar_W1, pvar_W2, pvar_W3, pvar_b1, pvar_b2, pvar_b3,
         pvar_bt1, pvar_bt2, pvar_g1, pvar_g2 };

  char* ws = (char*)d_ws;
  size_t off = 0;
  auto alloc = [&](size_t bytes) -> void* {
    void* p = ws + off;
    off = (off + bytes + 255) & ~(size_t)255;
    return p;
  };
  // bf16 transposed weights
  __bf16* wt_inW   = (__bf16*)alloc(128 * 32 * 2);
  __bf16* wt_edgeW = (__bf16*)alloc(128 * 32 * 2);
  __bf16* wt_c1Wl  = (__bf16*)alloc(128 * 128 * 2);
  __bf16* wt_c1Wr  = (__bf16*)alloc(128 * 128 * 2);
  __bf16* wt_c1We  = (__bf16*)alloc(128 * 128 * 2);
  __bf16* wt_c2Wl  = (__bf16*)alloc(128 * 128 * 2);
  __bf16* wt_c2Wr  = (__bf16*)alloc(128 * 128 * 2);
  __bf16* wt_c2We  = (__bf16*)alloc(128 * 128 * 2);
  __bf16* wt_oW1   = (__bf16*)alloc(128 * 128 * 2);
  __bf16* wt_oW2   = (__bf16*)alloc(128 * 128 * 2);
  __bf16* wt_cW1   = (__bf16*)alloc(256 * 224 * 2);
  __bf16* wt_cW2   = (__bf16*)alloc(256 * 256 * 2);
  __bf16* wt_cW3   = (__bf16*)alloc(64 * 256 * 2);
  __bf16* wt_vW1   = (__bf16*)alloc(256 * 224 * 2);
  __bf16* wt_vW2   = (__bf16*)alloc(256 * 256 * 2);
  __bf16* wt_vW3   = (__bf16*)alloc(64 * 256 * 2);
  // f32 activations / scratch
  float* xln   = (float*)alloc((size_t)KNODES * 32 * 4);
  float* x     = (float*)alloc((size_t)KNODES * 128 * 4);
  float* eemb  = (float*)alloc((size_t)NEDGE * 128 * 4);
  float* xl    = (float*)alloc((size_t)KNODES * 128 * 4);
  float* xr    = (float*)alloc((size_t)KNODES * 128 * 4);
  float* ef    = (float*)alloc((size_t)NEDGE * 128 * 4);
  float* logitb = (float*)alloc((size_t)2 * NEDGE * 4 * 4);
  float* exb    = (float*)alloc((size_t)2 * NEDGE * 4 * 4);
  unsigned* mxkey = (unsigned*)alloc((size_t)KNODES * 4 * 4);
  float* den   = (float*)alloc((size_t)KNODES * 4 * 4);
  float* agg   = (float*)alloc((size_t)KNODES * 128 * 4);
  float* h1    = (float*)alloc((size_t)KNODES * 128 * 4);
  float* h2    = (float*)alloc((size_t)KNODES * 128 * 4);
  float* ztmp  = (float*)alloc((size_t)KNODES * 128 * 4);
  float* zb    = (float*)alloc((size_t)KNODES * 128 * 4);

  auto T = [&](const float* W, __bf16* Wt, int K, int N, int Kp) {
    int tot = N * Kp;
    wtrans_kernel<<<dim3((tot + 255) / 256), dim3(256), 0, stream>>>(W, Wt, K, N, Kp);
  };
  auto GEMM = [&](const float* A, int M, int K, const __bf16* Wt, int Kp,
                  const float* bias, float* out, int N, int act) {
    gemm_rows16<<<dim3(M / 16), dim3(32), 0, stream>>>(A, M, K, (const uint32_t*)Wt, Kp,
                                                       bias, out, N, act);
  };

  // weight prep
  T(P[pin_W],   wt_inW,   32, 128, 32);
  T(P[pedge_W], wt_edgeW, 32, 128, 32);
  T(P[pc1_Wl], wt_c1Wl, 128, 128, 128);
  T(P[pc1_Wr], wt_c1Wr, 128, 128, 128);
  T(P[pc1_We], wt_c1We, 128, 128, 128);
  T(P[pc2_Wl], wt_c2Wl, 128, 128, 128);
  T(P[pc2_Wr], wt_c2Wr, 128, 128, 128);
  T(P[pc2_We], wt_c2We, 128, 128, 128);
  T(P[pout_W1], wt_oW1, 128, 128, 128);
  T(P[pout_W2], wt_oW2, 128, 128, 128);
  T(P[pcons_W1], wt_cW1, 208, 256, 224);
  T(P[pcons_W2], wt_cW2, 256, 256, 256);
  T(P[pcons_W3], wt_cW3, 256, 64, 256);
  T(P[pvar_W1],  wt_vW1, 208, 256, 224);
  T(P[pvar_W2],  wt_vW2, 256, 256, 256);
  T(P[pvar_W3],  wt_vW3, 256, 64, 256);

  // node / edge embeddings
  ln32_kernel<<<dim3(KNODES), dim3(32), 0, stream>>>(block_features, P[pin_ln_g], P[pin_ln_b], xln);
  GEMM(xln, KNODES, 32, wt_inW, 32, P[pin_b], x, 128, 1);
  GEMM(edge_attr, NEDGE, 32, wt_edgeW, 32, P[pedge_b], eemb, 128, 1);

  auto gat_layer = [&](const float* hin, __bf16* wl, __bf16* wr, __bf16* we,
                       const float* bl, const float* br, const float* att,
                       const float* cbias, const float* ng, const float* nb, float* hout) {
    GEMM(hin, KNODES, 128, wl, 128, bl, xl, 128, 0);
    GEMM(hin, KNODES, 128, wr, 128, br, xr, 128, 0);
    GEMM(eemb, NEDGE, 128, we, 128, (const float*)nullptr, ef, 128, 0);
    hipMemsetAsync(mxkey, 0, (size_t)KNODES * 4 * 4, stream);  // key 0 == "below -inf"
    hipMemsetAsync(den, 0, (size_t)KNODES * 4 * 4, stream);
    hipMemsetAsync(agg, 0, (size_t)KNODES * 128 * 4, stream);
    edge_logit_kernel<<<dim3(2 * NEDGE * 4 * 32 / 256), dim3(256), 0, stream>>>(
        ei, xl, xr, ef, att, logitb, mxkey);
    edge_exp_kernel<<<dim3((2 * NEDGE * 4 + 255) / 256), dim3(256), 0, stream>>>(
        ei, logitb, mxkey, exb, den);
    edge_agg_kernel<<<dim3(2 * NEDGE * 128 / 256), dim3(256), 0, stream>>>(
        ei, exb, den, xl, agg);
    ln_resid_kernel<<<dim3(KNODES), dim3(128), 0, stream>>>(agg, cbias, hin, ng, nb, hout);
  };
  gat_layer(x,  wt_c1Wl, wt_c1Wr, wt_c1We, P[pc1_bl], P[pc1_br], P[pc1_att],
            P[pc1_bias], P[pn1_g], P[pn1_b], h1);
  gat_layer(h1, wt_c2Wl, wt_c2Wr, wt_c2We, P[pc2_bl], P[pc2_br], P[pc2_att],
            P[pc2_bias], P[pn2_g], P[pn2_b], h2);

  // z head
  GEMM(h2,   KNODES, 128, wt_oW1, 128, P[pout_b1], ztmp, 128, 1);
  GEMM(ztmp, KNODES, 128, wt_oW2, 128, P[pout_b2], zb,   128, 0);

  // composers (the big WMMA work): 64 rows/block, 2 waves, 96KB dynamic LDS
  const size_t smem_bytes = 2 * 32 * 256 * 2 + 2 * 32 * 256 * 4;  // 96 KB
  float* outp = (float*)d_out;
  composer_kernel<<<dim3((NCONS + 63) / 64), dim3(64), smem_bytes, stream>>>(
      NCONS, c_sub, cons_bid, c_bnd, zb,
      (const uint32_t*)wt_cW1, P[pcons_b1], P[pcons_g1], P[pcons_bt1],
      (const uint32_t*)wt_cW2, P[pcons_b2], P[pcons_g2], P[pcons_bt2],
      (const uint32_t*)wt_cW3, P[pcons_b3], outp);
  composer_kernel<<<dim3((NVARS + 63) / 64), dim3(64), smem_bytes, stream>>>(
      NVARS, v_sub, vars_bid, v_bnd, zb,
      (const uint32_t*)wt_vW1, P[pvar_b1], P[pvar_g1], P[pvar_bt1],
      (const uint32_t*)wt_vW2, P[pvar_b2], P[pvar_g2], P[pvar_bt2],
      (const uint32_t*)wt_vW3, P[pvar_b3], outp + (size_t)NCONS * 64);
}
